// ContrastiveLoss_12017318494682
// MI455X (gfx1250) — matrix operand order repaired and checked by reading, not compiled, not run on hardware
//
#include <hip/hip_runtime.h>

// ContrastiveLoss for MI455X (gfx1250, wave32, WMMA).
// S = F_norm * F_norm^T fused with the contrastive epilogue; the 64MB sim
// matrix is never materialized. Compute-bound (AI ~2000 FLOP/B) -> f16 WMMA
// with f32 accumulation. B tiles are staged into LDS with CDNA5 async
// global->LDS copies (double-buffered, ASYNCcnt + barrier synchronized) and
// shared by all 8 waves of the block. Deterministic: per-(chunk,row)
// partials, no atomics.

typedef __attribute__((ext_vector_type(16))) _Float16 v16h;
typedef __attribute__((ext_vector_type(8)))  _Float16 v8h;
typedef __attribute__((ext_vector_type(8)))  float    v8f;
typedef __attribute__((ext_vector_type(4)))  int      v4i;

#define NROW 4096
#define NDIM 512
#define NCHUNK 16                        // column chunks
#define CHUNK_COLS (NROW / NCHUNK)       // 256 columns per chunk
#define CT_PER_CHUNK (CHUNK_COLS / 16)   // 16 column-tiles per chunk
#define LDS_STRIDE 520                   // halfs per LDS B row (512 + 8 pad)
                                         // -> lane lh hits banks 4*lh..4*lh+3

// ---------------------------------------------------------------------------
// CDNA5 async global->LDS 16B copy (ASYNCcnt-tracked) + wait helper.
// Builtin prototype (probed): (v4i addrspace(1)*, v4i addrspace(3)*, imm, imm)
// ---------------------------------------------------------------------------
__device__ __forceinline__ void async_cp16(const _Float16* gp, _Float16* lp) {
#if __has_builtin(__builtin_amdgcn_global_load_async_to_lds_b128)
  __builtin_amdgcn_global_load_async_to_lds_b128(
      (__attribute__((address_space(1))) v4i*)gp,
      (__attribute__((address_space(3))) v4i*)lp, 0, 0);
#else
  const unsigned ldsOff =
      (unsigned)(size_t)(__attribute__((address_space(3))) _Float16*)lp;
  asm volatile("global_load_async_to_lds_b128 %0, %1, off"
               :: "v"(ldsOff), "v"((unsigned long long)(size_t)gp)
               : "memory");
#endif
}

__device__ __forceinline__ void wait_async0() {
#if __has_builtin(__builtin_amdgcn_s_wait_asynccnt)
  __builtin_amdgcn_s_wait_asynccnt(0);
#else
  asm volatile("s_wait_asynccnt 0x0" ::: "memory");
#endif
}

// Stage one 16-row x 512-half B tile (rows nB..nB+15 of Fh) into LDS with
// padded row stride. 256 threads x 4 passes x 16B = 16KB.
__device__ __forceinline__ void stage_B_tile(const _Float16* __restrict__ Fh,
                                             int nB, _Float16* buf) {
  const int tid = threadIdx.x;
#pragma unroll
  for (int q = 0; q < 4; ++q) {
    const int lin  = q * 256 + tid;   // 16B granule index within tile
    const int row  = lin >> 6;        // 64 granules per 1KB row
    const int colh = (lin & 63) * 8;  // half index within row
    async_cp16(Fh + (size_t)(nB + row) * NDIM + colh,
               buf + row * LDS_STRIDE + colh);
  }
}

// ---------------------------------------------------------------------------
// Kernel 1: row L2-normalize (eps-clamped) f32 -> f16.  One wave per row.
// ---------------------------------------------------------------------------
__global__ __launch_bounds__(256) void cl_normalize(
    const float* __restrict__ F, _Float16* __restrict__ Fh) {
  const int wave = threadIdx.x >> 5;
  const int lane = threadIdx.x & 31;
  const int row  = blockIdx.x * 8 + wave;

  const float* src = F + (size_t)row * NDIM + lane * 16;
  float vals[16];
  float ss = 0.0f;
#pragma unroll
  for (int t = 0; t < 4; ++t) {
    float4 v = ((const float4*)src)[t];
    vals[4 * t + 0] = v.x; vals[4 * t + 1] = v.y;
    vals[4 * t + 2] = v.z; vals[4 * t + 3] = v.w;
    ss += v.x * v.x + v.y * v.y + v.z * v.z + v.w * v.w;
  }
#pragma unroll
  for (int m = 16; m >= 1; m >>= 1) ss += __shfl_xor(ss, m, 32);

  const float inv = 1.0f / fmaxf(sqrtf(ss), 1e-8f);

  v8h o0, o1;
#pragma unroll
  for (int t = 0; t < 8; ++t) {
    o0[t] = (_Float16)(vals[t] * inv);
    o1[t] = (_Float16)(vals[8 + t] * inv);
  }
  _Float16* dst = Fh + (size_t)row * NDIM + lane * 16;
  *(v8h*)dst       = o0;
  *(v8h*)(dst + 8) = o1;
}

// ---------------------------------------------------------------------------
// Kernel 2: WMMA Gram tiles + fused contrastive epilogue.
// Block = 8 waves sharing one 256-column chunk; each wave owns a 16-row
// strip. B tiles staged to LDS via async copies (double buffered); A strip
// pinned in 128 VGPRs; 16 K-steps of v_wmma_f32_16x16x32_f16 per tile.
// ---------------------------------------------------------------------------
__global__ __launch_bounds__(256) void cl_sim_wmma(
    const _Float16* __restrict__ Fh, const int* __restrict__ lab,
    float* __restrict__ pNeg, float* __restrict__ pPos,
    float* __restrict__ pCnt) {
  __shared__ __align__(32) _Float16 Bbuf[2][16 * LDS_STRIDE];

  const int lane  = threadIdx.x & 31;
  const int wave  = threadIdx.x >> 5;
  const int chunk = blockIdx.x & (NCHUNK - 1);     // all waves share chunk
  const int strip = (blockIdx.x >> 4) * 8 + wave;  // per-wave 16-row strip
  const int rowBase = strip * 16;
  const int colBase = chunk * CHUNK_COLS;
  const int lh = lane & 15;
  const int hi = lane >> 4;

  // Pin this wave's A strip (16 rows x 512 halfs) in registers, in a contiguous
  // k order (same as B): elems 0..15 = K(kk+16*hi..kk+16*hi+15).
  const _Float16* Arow = Fh + (size_t)(rowBase + lh) * NDIM;
  v16h Areg[16];
#pragma unroll
  for (int ks = 0; ks < 16; ++ks) {
    const int kk = ks * 32;
    v8h a0 = *(const v8h*)(Arow + kk + hi * 16);
    v8h a1 = *(const v8h*)(Arow + kk + hi * 16 + 8);
    Areg[ks] = __builtin_shufflevector(a0, a1, 0, 1, 2, 3, 4, 5, 6, 7,
                                       8, 9, 10, 11, 12, 13, 14, 15);
  }

  int li[8];
#pragma unroll
  for (int r = 0; r < 8; ++r) li[r] = lab[rowBase + hi * 8 + r];

  float accNeg[8] = {0.f, 0.f, 0.f, 0.f, 0.f, 0.f, 0.f, 0.f};
  float accPos[8] = {0.f, 0.f, 0.f, 0.f, 0.f, 0.f, 0.f, 0.f};
  float accCnt[8] = {0.f, 0.f, 0.f, 0.f, 0.f, 0.f, 0.f, 0.f};

  // Prologue: stage first B tile.
  stage_B_tile(Fh, colBase, &Bbuf[0][0]);
  wait_async0();
  __syncthreads();

  for (int ct = 0; ct < CT_PER_CHUNK; ++ct) {
    const int p  = ct & 1;
    const int nB = colBase + ct * 16;

    // Overlap: stage next tile into the other buffer while computing.
    if (ct + 1 < CT_PER_CHUNK)
      stage_B_tile(Fh, nB + 16, &Bbuf[p ^ 1][0]);

    // B operand from LDS: lane's column (nB + lh) is LDS row lh; 16
    // contiguous halfs starting at kk + 16*hi (same contiguous k order as A).
    const _Float16* Brow = &Bbuf[p][0] + lh * LDS_STRIDE + hi * 16;

    v8f acc = {};
#pragma unroll
    for (int ks = 0; ks < 16; ++ks) {
      const int kk = ks * 32;
      v8h b0 = *(const v8h*)(Brow + kk);
      v8h b1 = *(const v8h*)(Brow + kk + 8);
      v16h B = __builtin_shufflevector(b0, b1, 0, 1, 2, 3, 4, 5, 6, 7,
                                       8, 9, 10, 11, 12, 13, 14, 15);
      acc = __builtin_amdgcn_wmma_f32_16x16x32_f16(
          /*neg_a=*/false, Areg[ks], /*neg_b=*/false, B,
          /*c_mod=*/(short)0, acc, /*reuse_a=*/false, /*reuse_b=*/false);
    }

    // Fused epilogue on this 16x16 tile.
    const int j  = nB + lh;
    const int jl = lab[j];
#pragma unroll
    for (int r = 0; r < 8; ++r) {
      const int   i    = rowBase + hi * 8 + r;
      const float twoS = 2.0f * acc[r];          // sim / TEMPERATURE
      const float e    = __expf(twoS);
      const bool  diag = (i == j);
      const bool  same = (li[r] == jl);
      accNeg[r] += (!diag && !same) ? e    : 0.0f;
      accPos[r] += (!diag &&  same) ? twoS : 0.0f;
      accCnt[r] += (!diag &&  same) ? 1.0f : 0.0f;
    }

    // Ensure next tile staged AND all waves done reading before overwrite.
    wait_async0();
    __syncthreads();
  }

  // Reduce across the 16 lanes holding the same row, store partials once.
#pragma unroll
  for (int r = 0; r < 8; ++r) {
    float n = accNeg[r], p2 = accPos[r], c = accCnt[r];
#pragma unroll
    for (int m = 1; m < 16; m <<= 1) {
      n  += __shfl_xor(n, m, 32);
      p2 += __shfl_xor(p2, m, 32);
      c  += __shfl_xor(c, m, 32);
    }
    if (lh == 0) {
      const int idx = chunk * NROW + rowBase + hi * 8 + r;
      pNeg[idx] = n;
      pPos[idx] = p2;
      pCnt[idx] = c;
    }
  }
}

// ---------------------------------------------------------------------------
// Kernel 3: fixed-order chunk reduction, per-row loss, global reduction.
// loss_i = cnt_i * log(negsum_i) - sum_pos(2s);  out = sum / (nnz + 1e-5).
// ---------------------------------------------------------------------------
__global__ __launch_bounds__(256) void cl_finalize(
    const float* __restrict__ pNeg, const float* __restrict__ pPos,
    const float* __restrict__ pCnt, float* __restrict__ out) {
  __shared__ float sL[256];
  __shared__ float sC[256];
  float L = 0.0f, C = 0.0f;
  for (int i = threadIdx.x; i < NROW; i += 256) {
    float neg = 0.0f, pos = 0.0f, cnt = 0.0f;
    for (int c = 0; c < NCHUNK; ++c) {
      neg += pNeg[c * NROW + i];
      pos += pPos[c * NROW + i];
      cnt += pCnt[c * NROW + i];
    }
    if (cnt > 0.0f) {
      L += cnt * logf(neg) - pos;
      C += cnt;
    }
  }
  sL[threadIdx.x] = L;
  sC[threadIdx.x] = C;
  __syncthreads();
  for (int s = 128; s > 0; s >>= 1) {
    if ((int)threadIdx.x < s) {
      sL[threadIdx.x] += sL[threadIdx.x + s];
      sC[threadIdx.x] += sC[threadIdx.x + s];
    }
    __syncthreads();
  }
  if (threadIdx.x == 0) out[0] = sL[0] / (sC[0] + 1e-5f);
}

// ---------------------------------------------------------------------------
extern "C" void kernel_launch(void* const* d_in, const int* in_sizes, int n_in,
                              void* d_out, int out_size, void* d_ws,
                              size_t ws_size, hipStream_t stream) {
  (void)in_sizes; (void)n_in; (void)out_size; (void)ws_size;
  const float* F   = (const float*)d_in[0];
  const int*   lab = (const int*)d_in[1];
  float*       out = (float*)d_out;

  // Workspace layout: f16 matrix (4MB), then 3 partial arrays (NCHUNK*NROW).
  _Float16* Fh   = (_Float16*)d_ws;
  float*    pNeg = (float*)((char*)d_ws + (size_t)NROW * NDIM * sizeof(_Float16));
  float*    pPos = pNeg + (size_t)NCHUNK * NROW;
  float*    pCnt = pPos + (size_t)NCHUNK * NROW;

  cl_normalize<<<NROW / 8, 256, 0, stream>>>(F, Fh);
  // 512 blocks: 32 strip-groups x 16 chunks; 8 waves (strips) per block.
  cl_sim_wmma<<<(NROW / 16) / 8 * NCHUNK, 256, 0, stream>>>(Fh, lab, pNeg,
                                                            pPos, pCnt);
  cl_finalize<<<1, 256, 0, stream>>>(pNeg, pPos, pCnt, out);
}
